// GatedGCNNet_29781303231223
// MI455X (gfx1250) — matrix operand order repaired and checked by reading, not compiled, not run on hardware
//
#include <hip/hip_runtime.h>
#include <hip/hip_bf16.h>
#include <stdint.h>

// ---------------------------------------------------------------------------
// GatedGCNNet forward for MI455X (gfx1250, wave32, WMMA bf16 16x16x32).
// G=16 graphs, N=128 nodes, H=96 hidden, L=4 layers, NC=6 classes.
// ---------------------------------------------------------------------------

#define G 16
#define N 128
#define H 96
#define NC 6
#define L 4
#define GN (G * N)            // 2048
#define NH (N * H)            // 12288
#define EDGE_ELEMS ((size_t)G * N * N * H)   // 25,165,824

typedef __bf16 bf16;
typedef __attribute__((ext_vector_type(16))) __bf16 v16bf;
typedef __attribute__((ext_vector_type(8)))  float  v8f;

// ---------------- WMMA fragment helpers (wave32) ---------------------------
// A (16x32 bf16 from row-major f32): lanes 0-15 hold M, lane>>4 selects the
// K-octet; halves 0..7 <- K kh+0..7, halves 8..15 <- K kh+16..23.
// Two contiguous 8-float runs -> 4x global_load_b128 + packed cvt.
__device__ __forceinline__ v16bf load_a_frag_f32(const float* __restrict__ A, int lda,
                                                 int mbase, int kbase, int lane) {
  int m  = mbase + (lane & 15);
  int kh = (lane >> 4) * 8;
  const float* p = A + (size_t)m * lda + kbase + kh;
  float4 x0 = *(const float4*)(p);
  float4 x1 = *(const float4*)(p + 4);
  float4 y0 = *(const float4*)(p + 16);
  float4 y1 = *(const float4*)(p + 20);
  v16bf a;
  a[0]  = (bf16)x0.x; a[1]  = (bf16)x0.y; a[2]  = (bf16)x0.z; a[3]  = (bf16)x0.w;
  a[4]  = (bf16)x1.x; a[5]  = (bf16)x1.y; a[6]  = (bf16)x1.z; a[7]  = (bf16)x1.w;
  a[8]  = (bf16)y0.x; a[9]  = (bf16)y0.y; a[10] = (bf16)y0.z; a[11] = (bf16)y0.w;
  a[12] = (bf16)y1.x; a[13] = (bf16)y1.y; a[14] = (bf16)y1.z; a[15] = (bf16)y1.w;
  return a;
}

// Stage a 96x96 f32 B block (column offset ncol0, leading dim ldb) into LDS,
// bf16-converted and swizzled to the per-lane fragment layout:
//   Bpack[kb(3)][ct(6)][lane(32)][j(16)] = B[kb*32 + (lane>>4)*8 + (j&7) + (j>>3)*16][ct*16 + (lane&15)]
// so a B fragment is one aligned 32-byte LDS vector read.
__device__ __forceinline__ void stage_b_pack(const float* __restrict__ B, int ldb,
                                             int ncol0, bf16* Bpack, int tid) {
  for (int idx = tid; idx < 96 * 96; idx += 256) {
    int k = idx / 96, n = idx % 96;
    int kb = k >> 5, rem = k & 31;
    int lane = ((rem >> 3) & 1) * 16 + (n & 15);
    int j = (rem & 7) + ((rem >> 4) << 3);
    int ct = n >> 4;
    Bpack[(((kb * 6 + ct) * 32) + lane) * 16 + j] = (bf16)B[(size_t)k * ldb + ncol0 + n];
  }
}

__device__ __forceinline__ v16bf load_b_frag_lds(const bf16* Bpack, int kb, int ct,
                                                 int lane) {
  return *(const v16bf*)(Bpack + (((kb * 6 + ct) * 32) + lane) * 16);
}

__device__ __forceinline__ v8f wmma_bf16(v16bf a, v16bf b, v8f c) {
  return __builtin_amdgcn_wmma_f32_16x16x32_bf16(false, a, false, b, (short)0, c,
                                                 false, false);
}

// ---------------- Generic GEMM: C[M,Ntot] = A[M,K=96] @ B[96,Ntot] + bias ---
// grid = (M/128, Ntot/96), block = 256 (8 waves); each wave: one 16-row stripe
// x six 16-col tiles. f32 in, bf16 WMMA, f32 out. B staged once in LDS.
__global__ void gemm_bf16_bias(const float* __restrict__ A, int lda,
                               const float* __restrict__ B, int ldb,
                               const float* __restrict__ bias,
                               float* __restrict__ C, int ldc) {
  __shared__ bf16 Bpack[3 * 6 * 32 * 16];   // 18 KB
  int tid = threadIdx.x, lane = tid & 31, wave = tid >> 5;
  int mbase = blockIdx.x * 128 + wave * 16;
  int nblk  = blockIdx.y * 96;
  stage_b_pack(B, ldb, nblk, Bpack, tid);
  __syncthreads();

  v8f acc[6];
#pragma unroll
  for (int t = 0; t < 6; ++t)
#pragma unroll
    for (int r = 0; r < 8; ++r) acc[t][r] = 0.f;

#pragma unroll
  for (int kb = 0; kb < 3; ++kb) {
    v16bf a = load_a_frag_f32(A, lda, mbase, kb * 32, lane);
#pragma unroll
    for (int t = 0; t < 6; ++t) {
      v16bf b = load_b_frag_lds(Bpack, kb, t, lane);
      acc[t] = wmma_bf16(a, b, acc[t]);
    }
  }
#pragma unroll
  for (int t = 0; t < 6; ++t) {
    int n  = nblk + t * 16 + (lane & 15);
    int mo = mbase + (lane >> 4) * 8;
    float bv = bias[n];
#pragma unroll
    for (int r = 0; r < 8; ++r) C[(size_t)(mo + r) * ldc + n] = acc[t][r] + bv;
  }
}

// ---------------- Embedding ------------------------------------------------
__global__ void embed_kernel(const int* __restrict__ feat,
                             const float* __restrict__ emb,
                             float* __restrict__ h) {
  int node = blockIdx.x;
  int c = threadIdx.x;
  if (c < H) h[(size_t)node * H + c] = emb[(size_t)feat[node] * H + c];
}

// ---------------- Scorer: p2 = relu(BN(hs @ W_g2)) -------------------------
// One block per graph. WMMA accumulators stay in registers; per-column BN
// stats built from per-lane partial sums via a small LDS array.
__global__ void score_p2_kernel(const float* __restrict__ h,
                                const float* __restrict__ Wg2,
                                const float* __restrict__ gam,
                                const float* __restrict__ bet,
                                float* __restrict__ p2) {
  __shared__ bf16 Bpack[3 * 6 * 32 * 16];
  __shared__ float psum[16][H], psq[16][H];
  __shared__ float mean_s[H], inv_s[H];
  int g = blockIdx.x;
  int tid = threadIdx.x, lane = tid & 31, wave = tid >> 5;
  const float* A = h + (size_t)g * NH;
  int mbase = wave * 16;
  int widx = wave * 2 + (lane >> 4);
  int nlow = lane & 15;

  stage_b_pack(Wg2, H, 0, Bpack, tid);
  __syncthreads();

  v8f acc[6];
#pragma unroll
  for (int t = 0; t < 6; ++t)
#pragma unroll
    for (int r = 0; r < 8; ++r) acc[t][r] = 0.f;
#pragma unroll
  for (int kb = 0; kb < 3; ++kb) {
    v16bf a = load_a_frag_f32(A, H, mbase, kb * 32, lane);
#pragma unroll
    for (int t = 0; t < 6; ++t)
      acc[t] = wmma_bf16(a, load_b_frag_lds(Bpack, kb, t, lane), acc[t]);
  }
#pragma unroll
  for (int t = 0; t < 6; ++t) {
    float s = 0.f, s2 = 0.f;
#pragma unroll
    for (int r = 0; r < 8; ++r) { float v = acc[t][r]; s += v; s2 += v * v; }
    psum[widx][t * 16 + nlow] = s;
    psq[widx][t * 16 + nlow]  = s2;
  }
  __syncthreads();
  if (tid < H) {
    float s = 0.f, s2 = 0.f;
#pragma unroll
    for (int w = 0; w < 16; ++w) { s += psum[w][tid]; s2 += psq[w][tid]; }
    float m = s / (float)N;
    mean_s[tid] = m;
    inv_s[tid]  = rsqrtf(s2 / (float)N - m * m + 1e-5f);
  }
  __syncthreads();
#pragma unroll
  for (int t = 0; t < 6; ++t) {
    int n  = t * 16 + nlow;
    int mo = mbase + (lane >> 4) * 8;
    float gm = gam[n] * inv_s[n], mn = mean_s[n], bt = bet[n];
#pragma unroll
    for (int r = 0; r < 8; ++r) {
      float v = gm * (acc[t][r] - mn) + bt;
      p2[(size_t)g * NH + (mo + r) * H + n] = fmaxf(v, 0.f);
    }
  }
}

// ---------------- Scorer: q partials (deterministic, register-resident) ----
// q[n,c] = sum_k Wpg[k]*relu(BN(hs@Wg1)[n, k*H+c]); block (kg,g) handles 16
// k's, accumulating in registers, and writes one qpart slice.
__global__ void score_q_kernel(const float* __restrict__ h,
                               const float* __restrict__ Wg1,
                               const float* __restrict__ gam,
                               const float* __restrict__ bet,
                               const float* __restrict__ Wpg,
                               float* __restrict__ qpart) {
  __shared__ bf16 Bpack[3 * 6 * 32 * 16];
  __shared__ float psum[16][H], psq[16][H];
  __shared__ float mean_s[H], inv_s[H];
  int kg = blockIdx.x, g = blockIdx.y;
  int tid = threadIdx.x, lane = tid & 31, wave = tid >> 5;
  const float* A = h + (size_t)g * NH;
  float* qp = qpart + ((size_t)kg * G + g) * NH;
  int mbase = wave * 16;
  int widx = wave * 2 + (lane >> 4);
  int nlow = lane & 15;

  // A fragments are k-invariant: hoist out of the 16-iteration loop.
  v16bf afr[3];
#pragma unroll
  for (int kb = 0; kb < 3; ++kb) afr[kb] = load_a_frag_f32(A, H, mbase, kb * 32, lane);

  float qacc[6][8];
#pragma unroll
  for (int t = 0; t < 6; ++t)
#pragma unroll
    for (int r = 0; r < 8; ++r) qacc[t][r] = 0.f;

  for (int k = kg * 16; k < kg * 16 + 16; ++k) {
    stage_b_pack(Wg1, H * H, k * H, Bpack, tid);
    __syncthreads();
    v8f acc[6];
#pragma unroll
    for (int t = 0; t < 6; ++t)
#pragma unroll
      for (int r = 0; r < 8; ++r) acc[t][r] = 0.f;
#pragma unroll
    for (int kb = 0; kb < 3; ++kb)
#pragma unroll
      for (int t = 0; t < 6; ++t)
        acc[t] = wmma_bf16(afr[kb], load_b_frag_lds(Bpack, kb, t, lane), acc[t]);

#pragma unroll
    for (int t = 0; t < 6; ++t) {
      float s = 0.f, s2 = 0.f;
#pragma unroll
      for (int r = 0; r < 8; ++r) { float v = acc[t][r]; s += v; s2 += v * v; }
      psum[widx][t * 16 + nlow] = s;
      psq[widx][t * 16 + nlow]  = s2;
    }
    __syncthreads();
    if (tid < H) {
      float s = 0.f, s2 = 0.f;
#pragma unroll
      for (int w = 0; w < 16; ++w) { s += psum[w][tid]; s2 += psq[w][tid]; }
      float m = s / (float)N;
      mean_s[tid] = m;
      inv_s[tid]  = rsqrtf(s2 / (float)N - m * m + 1e-5f);
    }
    __syncthreads();
    float wk = Wpg[k];
#pragma unroll
    for (int t = 0; t < 6; ++t) {
      int n = t * 16 + nlow;
      float gm = gam[k * H + n] * inv_s[n], mn = mean_s[n], bt = bet[k * H + n];
#pragma unroll
      for (int r = 0; r < 8; ++r) {
        float v = gm * (acc[t][r] - mn) + bt;
        qacc[t][r] += wk * fmaxf(v, 0.f);
      }
    }
    __syncthreads();   // Bpack / stat arrays reused next k
  }
#pragma unroll
  for (int t = 0; t < 6; ++t) {
    int n  = t * 16 + nlow;
    int mo = mbase + (lane >> 4) * 8;
#pragma unroll
    for (int r = 0; r < 8; ++r) qp[(size_t)(mo + r) * H + n] = qacc[t][r];
  }
}

__global__ void q_reduce_kernel(const float* __restrict__ qpart, float* __restrict__ q) {
  int t = blockIdx.x * 256 + threadIdx.x;
  if (t >= GN * H) return;
  int g = t / NH, r = t % NH;
  float s = 0.f;
#pragma unroll
  for (int kg = 0; kg < 6; ++kg) s += qpart[((size_t)kg * G + g) * NH + r];
  q[t] = s;
}

// ---------------- Scorer: s = q @ p2^T + b_pg, zero diag -------------------
__global__ void score_s_kernel(const float* __restrict__ q,
                               const float* __restrict__ p2,
                               const float* __restrict__ b_pg,
                               float* __restrict__ s) {
  __shared__ float qi[H];
  int i = blockIdx.x, g = blockIdx.y, j = threadIdx.x;
  if (j < H) qi[j] = q[((size_t)g * N + i) * H + j];
  __syncthreads();
  const float* pr = p2 + ((size_t)g * N + j) * H;
  float acc = b_pg[0];
  for (int c = 0; c < H; ++c) acc += qi[c] * pr[c];
  s[((size_t)g * N + i) * N + j] = (i == j) ? 0.f : acc;
}

// ---------------- Mask: softmax(ax0)*softmax(ax1) > .5  | base, & ~eye -----
__global__ void mask_kernel(const float* __restrict__ s,
                            const float* __restrict__ adj,
                            float* __restrict__ maskf,
                            float* __restrict__ cntp) {
  __shared__ float rmax[N], rsum[N], cmax[N], csum[N], red[N];
  int g = blockIdx.x, t = threadIdx.x;
  const float* sg = s + (size_t)g * N * N;
  float mx = -1e30f;
  for (int j = 0; j < N; ++j) mx = fmaxf(mx, sg[t * N + j]);
  float sm = 0.f;
  for (int j = 0; j < N; ++j) sm += expf(sg[t * N + j] - mx);
  rmax[t] = mx; rsum[t] = sm;
  mx = -1e30f;
  for (int i = 0; i < N; ++i) mx = fmaxf(mx, sg[i * N + t]);
  sm = 0.f;
  for (int i = 0; i < N; ++i) sm += expf(sg[i * N + t] - mx);
  cmax[t] = mx; csum[t] = sm;
  __syncthreads();
  float cnt = 0.f;
  int i = t;
  for (int j = 0; j < N; ++j) {
    float v = sg[i * N + j];
    float mat = (expf(v - cmax[j]) / csum[j]) * (expf(v - rmax[i]) / rsum[i]);
    bool nw = mat > 0.5f;
    bool bs = adj[(size_t)g * N * N + i * N + j] < 0.125f;
    float m = ((nw || bs) && (i != j)) ? 1.f : 0.f;
    maskf[(size_t)g * N * N + i * N + j] = m;
    cnt += m;
  }
  red[t] = cnt;
  __syncthreads();
  for (int off = 64; off > 0; off >>= 1) {
    if (t < off) red[t] += red[t + off];
    __syncthreads();
  }
  if (t == 0) cntp[g] = red[0];
}

// ---------------- Edge init: e_pre = dist * W_e + b_e ----------------------
__global__ void edge_init_kernel(const float* __restrict__ h,
                                 const float* __restrict__ We,
                                 const float* __restrict__ be,
                                 float* __restrict__ e) {
  __shared__ float hi[H];
  __shared__ float dist[N];
  int gi = blockIdx.x;                 // g*N + i
  int g = gi >> 7, t = threadIdx.x;
  if (t < H) hi[t] = h[(size_t)gi * H + t];
  __syncthreads();
  {
    const float* hj = h + ((size_t)g * N + t) * H;
    float sm = 0.f;
    for (int c = 0; c < H; ++c) { float d = hi[c] - hj[c] + 1e-6f; sm += d * d; }
    dist[t] = sqrtf(sm);
  }
  __syncthreads();
  float* erow = e + (size_t)gi * NH;
  for (int idx = t; idx < NH; idx += N) {
    int j = idx / H, c = idx % H;
    erow[idx] = dist[j] * We[c] + be[c];
  }
}

// ---------------- Masked BN stats (per-block partials, deterministic) ------
__global__ void masked_stats_part(const float* __restrict__ x,
                                  const float* __restrict__ maskf,
                                  float* __restrict__ statp) {
  __shared__ float mrow[N];
  int gi = blockIdx.x;                 // g*N + i
  int t = threadIdx.x;
  mrow[t] = maskf[(size_t)gi * N + t];
  __syncthreads();
  if (t < H) {
    float s = 0.f, s2 = 0.f;
    const float* xr = x + (size_t)gi * NH;
    for (int j = 0; j < N; ++j) {
      float v = xr[j * H + t];
      float wv = mrow[j] * v;
      s += wv; s2 += wv * v;
    }
    statp[(size_t)gi * 192 + t]      = s;
    statp[(size_t)gi * 192 + 96 + t] = s2;
  }
}

__global__ void masked_stats_final(const float* __restrict__ statp,
                                   const float* __restrict__ cntp,
                                   float* __restrict__ stats) {
  __shared__ float csh;
  int t = threadIdx.x;
  if (t == 0) {
    float c = 0.f;
    for (int g = 0; g < G; ++g) c += cntp[g];
    csh = fmaxf(c, 1.f);
  }
  __syncthreads();
  if (t < H) {
    float s = 0.f, s2 = 0.f;
    for (int b = 0; b < GN; ++b) {
      s  += statp[(size_t)b * 192 + t];
      s2 += statp[(size_t)b * 192 + 96 + t];
    }
    float m = s / csh;
    stats[t]      = m;
    stats[96 + t] = rsqrtf(s2 / csh - m * m + 1e-5f);
  }
}

__global__ void bn_apply_kernel(const float* __restrict__ x,
                                const float* __restrict__ stats,
                                const float* __restrict__ gam,
                                const float* __restrict__ bet,
                                float* __restrict__ dst, int addTo) {
  size_t stride = (size_t)gridDim.x * blockDim.x;
  for (size_t idx = (size_t)blockIdx.x * blockDim.x + threadIdx.x; idx < EDGE_ELEMS;
       idx += stride) {
    int c = (int)(idx % H);
    float v = gam[c] * (x[idx] - stats[c]) * stats[96 + c] + bet[c];
    v = fmaxf(v, 0.f);
    dst[idx] = addTo ? dst[idx] + v : v;
  }
}

// ---------------- Layer: e_new += Dh+Eh, gated aggregation -----------------
__global__ void gate_aggregate_kernel(float* __restrict__ e_new,
                                      const float* __restrict__ Dh,
                                      const float* __restrict__ Eh,
                                      const float* __restrict__ Bh,
                                      const float* __restrict__ maskf,
                                      float* __restrict__ num,
                                      float* __restrict__ den) {
  int gj = blockIdx.x;                 // g*N + j  (destination node)
  int g = gj >> 7, j = gj & 127, c = threadIdx.x;
  if (c >= H) return;
  float Ehv = Eh[(size_t)gj * H + c];
  float nu = 0.f, de = 0.f;
  for (int i = 0; i < N; ++i) {
    size_t idx = (((size_t)g * N + i) * N + j) * H + c;
    float v = e_new[idx] + Dh[((size_t)g * N + i) * H + c] + Ehv;
    e_new[idx] = v;
    float w = maskf[((size_t)g * N + i) * N + j];
    float sg = w / (1.f + expf(-v));
    nu += sg * Bh[((size_t)g * N + i) * H + c];
    de += sg;
  }
  num[(size_t)gj * H + c] = nu;
  den[(size_t)gj * H + c] = de;
}

// ---------------- Layer: h += relu(BN(Ah + num/den)) over (G,N) ------------
__global__ void h_update_kernel(const float* __restrict__ Ah,
                                const float* __restrict__ num,
                                const float* __restrict__ den,
                                const float* __restrict__ gam,
                                const float* __restrict__ bet,
                                float* __restrict__ h) {
  __shared__ float rs[256], rs2[256];
  __shared__ float msh, ish;
  int c = blockIdx.x, t = threadIdx.x;
  float s = 0.f, s2 = 0.f;
  for (int n = t; n < GN; n += 256) {
    float x = Ah[(size_t)n * H + c] + num[(size_t)n * H + c] / (den[(size_t)n * H + c] + 1e-6f);
    s += x; s2 += x * x;
  }
  rs[t] = s; rs2[t] = s2;
  __syncthreads();
  for (int off = 128; off > 0; off >>= 1) {
    if (t < off) { rs[t] += rs[t + off]; rs2[t] += rs2[t + off]; }
    __syncthreads();
  }
  if (t == 0) {
    float m = rs[0] / (float)GN;
    msh = m;
    ish = rsqrtf(rs2[0] / (float)GN - m * m + 1e-5f);
  }
  __syncthreads();
  float gc = gam[c], bc = bet[c];
  for (int n = t; n < GN; n += 256) {
    float x = Ah[(size_t)n * H + c] + num[(size_t)n * H + c] / (den[(size_t)n * H + c] + 1e-6f);
    float v = fmaxf(gc * (x - msh) * ish + bc, 0.f);
    h[(size_t)n * H + c] += v;
  }
}

// ---------------- MLP readout ---------------------------------------------
__global__ void mlp_kernel(const float* __restrict__ h,
                           const float* __restrict__ W0, const float* __restrict__ b0,
                           const float* __restrict__ W1, const float* __restrict__ b1,
                           const float* __restrict__ W2, const float* __restrict__ b2,
                           float* __restrict__ out) {
  int n = blockIdx.x * 256 + threadIdx.x;
  if (n >= GN) return;
  const float* hr = h + (size_t)n * H;
  float t0[48], t1[24];
#pragma unroll
  for (int o = 0; o < 48; ++o) {
    float a = b0[o];
    for (int k = 0; k < 96; ++k) a += hr[k] * W0[k * 48 + o];
    t0[o] = fmaxf(a, 0.f);
  }
#pragma unroll
  for (int o = 0; o < 24; ++o) {
    float a = b1[o];
#pragma unroll
    for (int k = 0; k < 48; ++k) a += t0[k] * W1[k * 24 + o];
    t1[o] = fmaxf(a, 0.f);
  }
#pragma unroll
  for (int o = 0; o < NC; ++o) {
    float a = b2[o];
#pragma unroll
    for (int k = 0; k < 24; ++k) a += t1[k] * W2[k * NC + o];
    out[(size_t)n * NC + o] = a;
  }
}

// ---------------------------------------------------------------------------
// Host launcher. Input order = setup_inputs() dict insertion order:
// 0 feat(int) 1 adj_rand 2 emb_h 3 W_e 4 b_e 5 bn_lre_g 6 bn_lre_b 7 W_g1
// 8 b_g1 9 bn_g1_g 10 bn_g1_b 11 W_g2 12 b_g2 13 bn_g2_g 14 bn_g2_b 15 W_pg
// 16 b_pg, layers l: base 17+14l -> A bA B bB C bC D bD E bE bn_h_g bn_h_b
// bn_e_g bn_e_b, then 73..78 mlp_W0 b0 W1 b1 W2 b2.
// (b_g1/b_g2 are skipped: per-channel constants cancel inside BatchNorm.)
// ---------------------------------------------------------------------------
extern "C" void kernel_launch(void* const* d_in, const int* in_sizes, int n_in,
                              void* d_out, int out_size, void* d_ws, size_t ws_size,
                              hipStream_t stream) {
  const int*   feat = (const int*)d_in[0];
  const float* adj  = (const float*)d_in[1];
  const float* emb  = (const float*)d_in[2];
  const float* W_e  = (const float*)d_in[3];
  const float* b_e  = (const float*)d_in[4];
  const float* lre_g = (const float*)d_in[5];
  const float* lre_b = (const float*)d_in[6];
  const float* W_g1 = (const float*)d_in[7];
  const float* g1_g = (const float*)d_in[9];
  const float* g1_b = (const float*)d_in[10];
  const float* W_g2 = (const float*)d_in[11];
  const float* g2_g = (const float*)d_in[13];
  const float* g2_b = (const float*)d_in[14];
  const float* W_pg = (const float*)d_in[15];
  const float* b_pg = (const float*)d_in[16];
  const float* mlpW0 = (const float*)d_in[73];
  const float* mlpb0 = (const float*)d_in[74];
  const float* mlpW1 = (const float*)d_in[75];
  const float* mlpb1 = (const float*)d_in[76];
  const float* mlpW2 = (const float*)d_in[77];
  const float* mlpb2 = (const float*)d_in[78];

  float* ws = (float*)d_ws;
  float* h     = ws;                         // 196608
  float* e     = h + (size_t)GN * H;         // 25,165,824
  float* en    = e + EDGE_ELEMS;             // 25,165,824
  float* maskf = en + EDGE_ELEMS;            // 262144
  float* p2    = maskf + (size_t)G * N * N;  // 196608
  float* q     = p2 + (size_t)GN * H;        // 196608
  float* qpart = q + (size_t)GN * H;         // 1,179,648
  float* sbuf  = qpart + (size_t)6 * GN * H; // 262144
  float* Ah    = sbuf + (size_t)G * N * N;
  float* Bh    = Ah + (size_t)GN * H;
  float* Dh    = Bh + (size_t)GN * H;
  float* Ehb   = Dh + (size_t)GN * H;
  float* numb  = Ehb + (size_t)GN * H;
  float* denb  = numb + (size_t)GN * H;
  float* statp = denb + (size_t)GN * H;      // 2048*192
  float* stats = statp + (size_t)GN * 192;   // 192 (mean|invstd)
  float* cntp  = stats + 256;                // 16

  // 1) embedding
  embed_kernel<<<GN, 128, 0, stream>>>(feat, emb, h);

  // 2) edge scorer (per-graph batch stats), algebraically reduced to q @ p2^T
  score_p2_kernel<<<G, 256, 0, stream>>>(h, W_g2, g2_g, g2_b, p2);
  score_q_kernel<<<dim3(6, G), 256, 0, stream>>>(h, W_g1, g1_g, g1_b, W_pg, qpart);
  q_reduce_kernel<<<(GN * H + 255) / 256, 256, 0, stream>>>(qpart, q);
  score_s_kernel<<<dim3(N, G), N, 0, stream>>>(q, p2, b_pg, sbuf);
  mask_kernel<<<G, N, 0, stream>>>(sbuf, adj, maskf, cntp);

  // 3) edge init: e = relu(maskedBN(dist * W_e + b_e))
  edge_init_kernel<<<GN, N, 0, stream>>>(h, W_e, b_e, e);
  masked_stats_part<<<GN, N, 0, stream>>>(e, maskf, statp);
  masked_stats_final<<<1, 128, 0, stream>>>(statp, cntp, stats);
  bn_apply_kernel<<<8192, 256, 0, stream>>>(e, stats, lre_g, lre_b, e, 0);

  // 4) GatedGCN layers
  for (int l = 0; l < L; ++l) {
    int base = 17 + l * 14;
    const float* Wa = (const float*)d_in[base + 0];
    const float* ba = (const float*)d_in[base + 1];
    const float* Wb = (const float*)d_in[base + 2];
    const float* bb = (const float*)d_in[base + 3];
    const float* Wc = (const float*)d_in[base + 4];
    const float* bc = (const float*)d_in[base + 5];
    const float* Wd = (const float*)d_in[base + 6];
    const float* bd = (const float*)d_in[base + 7];
    const float* We2 = (const float*)d_in[base + 8];
    const float* be2 = (const float*)d_in[base + 9];
    const float* bnhg = (const float*)d_in[base + 10];
    const float* bnhb = (const float*)d_in[base + 11];
    const float* bneg = (const float*)d_in[base + 12];
    const float* bneb = (const float*)d_in[base + 13];

    // node projections (WMMA bf16)
    gemm_bf16_bias<<<dim3(GN / 128, 1), 256, 0, stream>>>(h, H, Wa, H, ba, Ah, H);
    gemm_bf16_bias<<<dim3(GN / 128, 1), 256, 0, stream>>>(h, H, Wb, H, bb, Bh, H);
    gemm_bf16_bias<<<dim3(GN / 128, 1), 256, 0, stream>>>(h, H, Wd, H, bd, Dh, H);
    gemm_bf16_bias<<<dim3(GN / 128, 1), 256, 0, stream>>>(h, H, We2, H, be2, Ehb, H);
    // edge projection: en = e @ C + bC   (M = G*N*N = 262144)
    gemm_bf16_bias<<<dim3((int)(EDGE_ELEMS / H / 128), 1), 256, 0, stream>>>(
        e, H, Wc, H, bc, en, H);
    // en += Dh_src + Eh_dst ; gated aggregation (num, den)
    gate_aggregate_kernel<<<GN, 128, 0, stream>>>(en, Dh, Ehb, Bh, maskf, numb, denb);
    // h += relu(BN(Ah + num/den))
    h_update_kernel<<<H, 256, 0, stream>>>(Ah, numb, denb, bnhg, bnhb, h);
    // e += relu(maskedBN(en))
    masked_stats_part<<<GN, N, 0, stream>>>(en, maskf, statp);
    masked_stats_final<<<1, 128, 0, stream>>>(statp, cntp, stats);
    bn_apply_kernel<<<8192, 256, 0, stream>>>(en, stats, bneg, bneb, e, 1);
  }

  // 5) MLP readout -> d_out [2048, 6] f32
  mlp_kernel<<<(GN + 255) / 256, 256, 0, stream>>>(h, mlpW0, mlpb0, mlpW1, mlpb1,
                                                   mlpW2, mlpb2, (float*)d_out);
}
